// BeyazKusAIAttention_36515811951168
// MI455X (gfx1250) — compile-verified, matrix operand order checked by reading, not run on hardware
//
#include <hip/hip_runtime.h>
#include <hip/hip_bf16.h>
#include <cstdint>
#include <cstddef>

// ---------------------------------------------------------------------------
// BeyazKusAIAttention collapses: softmax over a size-1 axis == 1.0, so
// attn==1, out = v broadcast over heads, and the whole op is
//   V = X @ Wv^T ;  Wos[d,r] = sum_h Wo[d, h*1024+r] ;  OUT = V @ Wos^T
// Wq / Wk / RoPE / mask are dead. Two f16-WMMA GEMMs with f32 accumulation,
// fed by gfx1250 async global->LDS DMA with double-buffered tiles.
// ---------------------------------------------------------------------------

typedef __attribute__((ext_vector_type(16))) _Float16 v16h;
typedef __attribute__((ext_vector_type(8)))  _Float16 v8h;
typedef __attribute__((ext_vector_type(8)))  float    v8f;

#define DIM_     4096
#define KV_RANK_ 1024
#define ROWS_    (4 * 4096)   // B * S token rows

#define BM 128
#define BN 128
#define BK 32
#define LDK (BK + 8)          // padded LDS stride (f16 elems) -> bank-conflict-free b128 reads

// gfx1250 async DMA: global -> LDS, 16B per lane, tracked by ASYNCcnt.
__device__ __forceinline__ void async_copy_b128(uint32_t lds_addr, const _Float16* gaddr) {
  asm volatile("global_load_async_to_lds_b128 %0, %1, off"
               :: "v"(lds_addr), "v"(gaddr)
               : "memory");
}
__device__ __forceinline__ void wait_async0() {
  asm volatile("s_wait_asynccnt 0x0" ::: "memory");
}

// ---- elementwise f32 -> f16 convert (grid-stride) -------------------------
__global__ void f32_to_f16_kernel(const float* __restrict__ in,
                                  _Float16* __restrict__ out, size_t n) {
  size_t stride = (size_t)gridDim.x * blockDim.x;
  for (size_t i = (size_t)blockIdx.x * blockDim.x + threadIdx.x; i < n; i += stride)
    out[i] = (_Float16)in[i];
}

// ---- fold Wo's 4 head-columns: out[d*1024+r] = sum_h Wo[d*4096 + h*1024 + r]
__global__ void wosum_f16_kernel(const float* __restrict__ Wo,
                                 _Float16* __restrict__ out) {
  const size_t n = (size_t)DIM_ * KV_RANK_;
  size_t stride = (size_t)gridDim.x * blockDim.x;
  for (size_t i = (size_t)blockIdx.x * blockDim.x + threadIdx.x; i < n; i += stride) {
    size_t d = i >> 10;               // / KV_RANK_
    size_t r = i & (KV_RANK_ - 1);
    const float* p = Wo + d * DIM_ + r;
    out[i] = (_Float16)(p[0] + p[KV_RANK_] + p[2 * KV_RANK_] + p[3 * KV_RANK_]);
  }
}

// ---- tiled GEMM: C[M,N] = A[M,K] * B[N,K]^T, f16 in, f32 accum ------------
// 256 threads = 8 wave32; waves tiled 4(M) x 2(N); each wave owns 32x64 of C
// (2x4 WMMA 16x16 tiles -> 8 v_wmma_f32_16x16x32_f16 per K step).
// Double-buffered LDS tiles filled by global_load_async_to_lds_b128:
//   wait(cur) -> barrier -> issue(next) -> compute(cur)
template <typename TOUT>
__global__ __launch_bounds__(256) void gemm_f16_wmma(
    const _Float16* __restrict__ A, const _Float16* __restrict__ B,
    TOUT* __restrict__ C, int M, int N, int K) {
  __shared__ _Float16 As[2][BM * LDK];
  __shared__ _Float16 Bs[2][BN * LDK];

  const int tid  = threadIdx.x;
  const int lane = tid & 31;
  const int wave = tid >> 5;     // 0..7
  const int wm   = wave >> 1;    // 0..3 : 32-row band
  const int wn   = wave & 1;     // 0..1 : 64-col band
  const int r    = lane & 15;    // row/col within 16x16 tile
  const int half = lane >> 4;    // K-half selector per ISA 16-bit A layout

  const int blockN = blockIdx.x * BN;
  const int blockM = blockIdx.y * BM;

  // Per-thread DMA slots: each thread moves two 16B vectors of A and of B per
  // K-tile. Row = tid>>2 (and +64), k-chunk = (tid&3)*8.
  const int mrow = tid >> 2;            // 0..63
  const int kvec = (tid & 3) * 8;       // 0,8,16,24

  const _Float16* gA0 = A + (size_t)(blockM + mrow) * K + kvec;
  const _Float16* gA1 = gA0 + (size_t)64 * K;
  const _Float16* gB0 = B + (size_t)(blockN + mrow) * K + kvec;
  const _Float16* gB1 = gB0 + (size_t)64 * K;

  // LDS aperture: low 32 bits of a generic shared pointer == LDS byte address.
  const uint32_t ldsA0 = (uint32_t)(uintptr_t)&As[0][mrow * LDK + kvec];
  const uint32_t ldsA1 = (uint32_t)(uintptr_t)&As[0][(mrow + 64) * LDK + kvec];
  const uint32_t ldsB0 = (uint32_t)(uintptr_t)&Bs[0][mrow * LDK + kvec];
  const uint32_t ldsB1 = (uint32_t)(uintptr_t)&Bs[0][(mrow + 64) * LDK + kvec];
  const uint32_t bufStride = (uint32_t)(BM * LDK * sizeof(_Float16));

  v8f acc[2][4];
#pragma unroll
  for (int tm = 0; tm < 2; ++tm)
#pragma unroll
    for (int tn = 0; tn < 4; ++tn)
#pragma unroll
      for (int j = 0; j < 8; ++j) acc[tm][tn][j] = 0.0f;

  // Prologue: start DMA of first K-tile into buffer 0.
  async_copy_b128(ldsA0, gA0);
  async_copy_b128(ldsA1, gA1);
  async_copy_b128(ldsB0, gB0);
  async_copy_b128(ldsB1, gB1);

  int buf = 0;
  for (int k0 = 0; k0 < K; k0 += BK, buf ^= 1) {
    wait_async0();        // our copies into As/Bs[buf] have landed
    __syncthreads();      // everyone's copies landed; prior reads of buf^1 done

    if (k0 + BK < K) {    // kick off DMA of next tile into the other buffer
      uint32_t bo = (buf ^ 1) ? bufStride : 0u;
      async_copy_b128(ldsA0 + bo, gA0 + k0 + BK);
      async_copy_b128(ldsA1 + bo, gA1 + k0 + BK);
      async_copy_b128(ldsB0 + bo, gB0 + k0 + BK);
      async_copy_b128(ldsB1 + bo, gB1 + k0 + BK);
    }

    const _Float16* Ab = As[buf];
    const _Float16* Bb = Bs[buf];

    // Fragment loads per CDNA5 16-bit A-matrix VGPR layout:
    // lanes 0-15 : K = {0..7, 16..23}; lanes 16-31 : K = {8..15, 24..31}.
    v16h af[2];
#pragma unroll
    for (int tm = 0; tm < 2; ++tm) {
      int row = wm * 32 + tm * 16 + r;
      v8h lo = *reinterpret_cast<const v8h*>(Ab + row * LDK + half * 8);
      v8h hi = *reinterpret_cast<const v8h*>(Ab + row * LDK + 16 + half * 8);
#pragma unroll
      for (int j = 0; j < 8; ++j) { af[tm][j] = lo[j]; af[tm][8 + j] = hi[j]; }
    }
    v16h bfr[4];
#pragma unroll
    for (int tn = 0; tn < 4; ++tn) {
      int row = wn * 64 + tn * 16 + r;
      v8h lo = *reinterpret_cast<const v8h*>(Bb + row * LDK + half * 8);
      v8h hi = *reinterpret_cast<const v8h*>(Bb + row * LDK + 16 + half * 8);
#pragma unroll
      for (int j = 0; j < 8; ++j) { bfr[tn][j] = lo[j]; bfr[tn][8 + j] = hi[j]; }
    }

#pragma unroll
    for (int tm = 0; tm < 2; ++tm)
#pragma unroll
      for (int tn = 0; tn < 4; ++tn)
        acc[tm][tn] = __builtin_amdgcn_wmma_f32_16x16x32_f16(
            false, af[tm], false, bfr[tn], (short)0, acc[tm][tn], false, false);
  }

  // C/D layout: VGPR j -> row j (lanes 0-15) / row j+8 (lanes 16-31), col = lane&15.
#pragma unroll
  for (int tm = 0; tm < 2; ++tm)
#pragma unroll
    for (int tn = 0; tn < 4; ++tn)
#pragma unroll
      for (int j = 0; j < 8; ++j) {
        int gr = blockM + wm * 32 + tm * 16 + half * 8 + j;
        int gc = blockN + wn * 64 + tn * 16 + r;
        C[(size_t)gr * N + gc] = (TOUT)acc[tm][tn][j];
      }
}

extern "C" void kernel_launch(void* const* d_in, const int* in_sizes, int n_in,
                              void* d_out, int out_size, void* d_ws, size_t ws_size,
                              hipStream_t stream) {
  (void)in_sizes; (void)n_in; (void)out_size; (void)ws_size;
  // setup_inputs order: x, Wq, Wk, Wv, Wo, mask — Wq/Wk/mask are dead (see header).
  const float* x  = (const float*)d_in[0];
  const float* Wv = (const float*)d_in[3];
  const float* Wo = (const float*)d_in[4];
  float* out = (float*)d_out;

  const size_t M    = ROWS_;                       // 16384 token rows
  const size_t nX   = M * DIM_;                    // 67.1M
  const size_t nWv  = (size_t)KV_RANK_ * DIM_;     // 4.2M
  const size_t nWos = (size_t)DIM_ * KV_RANK_;     // 4.2M
  // workspace layout (f16): Xh | Wvh | Wosh | Vh  (~184.5 MB total)
  char* ws = (char*)d_ws;
  _Float16* Xh   = (_Float16*)ws;  ws += nX * sizeof(_Float16);
  _Float16* Wvh  = (_Float16*)ws;  ws += nWv * sizeof(_Float16);
  _Float16* Wosh = (_Float16*)ws;  ws += nWos * sizeof(_Float16);
  _Float16* Vh   = (_Float16*)ws;  // M x KV_RANK_

  f32_to_f16_kernel<<<4096, 256, 0, stream>>>(x, Xh, nX);
  f32_to_f16_kernel<<<512, 256, 0, stream>>>(Wv, Wvh, nWv);
  wosum_f16_kernel<<<512, 256, 0, stream>>>(Wo, Wosh);

  // Stage 1: V = X @ Wv^T  (16384 x 1024, K = 4096), V kept f16 in L2-resident ws.
  dim3 g1(KV_RANK_ / BN, M / BM);                  // (8, 128)
  gemm_f16_wmma<_Float16><<<g1, 256, 0, stream>>>(Xh, Wvh, Vh,
                                                  (int)M, KV_RANK_, DIM_);
  // Stage 2: OUT = V @ Wos^T (16384 x 4096, K = 1024), f32 output.
  dim3 g2(DIM_ / BN, M / BM);                      // (32, 128)
  gemm_f16_wmma<float><<<g2, 256, 0, stream>>>(Vh, Wosh, out,
                                               (int)M, DIM_, KV_RANK_);
}
